// Temp_Causal_Address_38268158608009
// MI455X (gfx1250) — compile-verified, matrix operand order checked by reading, not run on hardware
//
#include <hip/hip_runtime.h>
#include <hip/hip_bf16.h>

// CDNA5 / gfx1250, wave32. WMMA used as a 32-wide cross-lane reduction engine:
// A = ones(16x32) f16  =>  D[m,n] = sum_k B[k,n]  (column sums of B).
// B columns carry one output row's softmax window each, so every lane's D[0]
// is the full k=32 reduction for its own row -- no extraction shuffles needed.

typedef __attribute__((ext_vector_type(16))) _Float16 v16h;
typedef __attribute__((ext_vector_type(8)))  float    v8f;

#define B_  8
#define E_  8
#define F_  16
#define T_  2048
#define K_  32
#define FO_ (F_ + 1)   // 17 output channels (16 encoded + endo pass-through)

__global__ __launch_bounds__(256)
void temp_causal_main(const float* __restrict__ endo,
                      const float* __restrict__ exo,
                      const float* __restrict__ w_expand,
                      const float* __restrict__ b_expand,
                      float* __restrict__ out)
{
    // blockIdx.x encodes (b, e, f) -> block-uniform, so w/bias are scalar loads.
    const int bef = blockIdx.x;
    const int f = bef % F_;
    const int e = (bef / F_) % E_;
    const int b = bef / (F_ * E_);

    const int lane  = threadIdx.x & 31;
    const int wave  = threadIdx.x >> 5;   // 0..7
    const int row   = lane & 15;          // t within 16-row tile
    const int jbase = (lane >> 4) << 4;   // 0 or 16: which half of the k=32 window

    const float w  = w_expand[e * F_ + f];
    const float bi = b_expand[e * F_ + f];

    const float* endoRow = endo + (b * E_ + e) * T_;
    const float* exoRow  = exo  + (b * F_ + f) * T_;
    float*       outRow  = out  + ((b * E_ + e) * FO_ + f) * T_;

    // Pull the whole exo row (8 KB) toward the caches: global_prefetch_b8 path.
    __builtin_prefetch(exoRow + lane * 64, 0, 1);

    // A operand: all-ones 16x32 f16 matrix (layout-independent since constant).
    v16h ones;
    #pragma unroll
    for (int i = 0; i < 16; ++i) ones[i] = (_Float16)1.0f;

    const int tilesPerWave = (T_ / 16) / 8;  // 16 tiles of 16 t-values per wave
    for (int it = 0; it < tilesPerWave; ++it) {
        const int t0 = (wave + it * 8) * 16;
        const int t  = t0 + row;

        const float q   = endoRow[t] * w + bi;
        const float aq  = fabsf(q);
        const float sgn = (q > 0.0f) ? 1.0f : ((q < 0.0f) ? -1.0f : 0.0f);

        // This lane's 16 contiguous taps of the causal window (zero-padded left edge).
        const int base = t - (K_ - 1) + jbase;   // base+15 <= t < T_, always in-range high
        float akv[16];
        float wmax = 0.0f;
        #pragma unroll
        for (int i = 0; i < 16; ++i) {
            int   src = base + i;
            float v   = exoRow[src >= 0 ? src : 0];
            v         = (src >= 0) ? v : 0.0f;
            akv[i]    = fabsf(v);
            wmax      = fmaxf(wmax, akv[i]);
        }
        // Full-row max over all 32 taps: other half of the window lives in lane^16.
        wmax = fmaxf(wmax, __shfl_xor(wmax, 16, 32));
        const float m = aq * wmax;               // softmax max (aq >= 0)

        // Build B operands (32x16 f16): lanes 0-15 carry K=0..15 of column N=lane,
        // lanes 16-31 carry K=16..31 of column N=lane-16; element i -> K = jbase+i.
        // e_j in (0,1], e_j*|kv_j| bounded -> safe in f16, accumulated in f32.
        v16h Be, Bev;
        #pragma unroll
        for (int i = 0; i < 16; ++i) {
            float ej = __expf(aq * akv[i] - m);
            Be[i]    = (_Float16)ej;
            Bev[i]   = (_Float16)(ej * akv[i]);
        }

        v8f c0 = {};
        // D[m,n] = sum_k B[k,n]: denominator and numerator of the softmax sum.
        v8f den = __builtin_amdgcn_wmma_f32_16x16x32_f16(
            false, ones, false, Be,  (short)0, c0, false, false);
        v8f num = __builtin_amdgcn_wmma_f32_16x16x32_f16(
            false, ones, false, Bev, (short)0, c0, false, false);

        // Every lane's d[0] is the full k-sum for its own row (N = lane & 15);
        // lanes 0-15 commit the 16 outputs of this tile.
        // den >= 1 always (max term contributes exp(0) = 1), so the raw hardware
        // v_rcp_f32 (~1 ULP) is safe and avoids the IEEE divide sequence.
        if (lane < 16) {
            outRow[t] = (sgn * num[0]) * __builtin_amdgcn_rcpf(den[0]);
        }
    }
}

// Pass-through channel: combined[:, :, 16, :] = endo
__global__ __launch_bounds__(256)
void temp_causal_copy(const float* __restrict__ endo, float* __restrict__ out)
{
    int i = blockIdx.x * blockDim.x + threadIdx.x;   // over B*E*T
    if (i < B_ * E_ * T_) {
        int t  = i % T_;
        int be = i / T_;
        out[(be * FO_ + F_) * T_ + t] = endo[i];
    }
}

extern "C" void kernel_launch(void* const* d_in, const int* in_sizes, int n_in,
                              void* d_out, int out_size, void* d_ws, size_t ws_size,
                              hipStream_t stream)
{
    (void)in_sizes; (void)n_in; (void)out_size; (void)d_ws; (void)ws_size;
    const float* endo = (const float*)d_in[0];   // (8, 8, 2048)  f32
    const float* exo  = (const float*)d_in[1];   // (8, 16, 2048) f32
    const float* wexp = (const float*)d_in[2];   // (8, 16)       f32
    const float* bexp = (const float*)d_in[3];   // (8, 16)       f32
    // d_in[4] = k_lookback (== 32, specialized at compile time)
    float* out = (float*)d_out;                  // (8, 8, 17, 2048) f32

    temp_causal_main<<<B_ * E_ * F_, 256, 0, stream>>>(endo, exo, wexp, bexp, out);

    const int n = B_ * E_ * T_;
    temp_causal_copy<<<(n + 255) / 256, 256, 0, stream>>>(endo, out);
}
